// SimpleAttentionFT_86139864088959
// MI455X (gfx1250) — matrix operand order repaired and checked by reading, not compiled, not run on hardware
//
#include <hip/hip_runtime.h>

// ---------------------------------------------------------------------------
// Types / helpers
// ---------------------------------------------------------------------------
typedef __attribute__((ext_vector_type(16))) _Float16 v16h;
typedef __attribute__((ext_vector_type(8)))  _Float16 v8h;
typedef __attribute__((ext_vector_type(8)))  float    v8f;
typedef __attribute__((ext_vector_type(4)))  float    v4f;

#define LDSROW 72                 // halfs per LDS tile row (64 data + 8 pad) -> 144B stride
#define TILEB  (128 * LDSROW * 2) // bytes per LDS tile buffer (18432)
#define L2E    1.44269504088896f

#define WMMA_F16(A, B, C) \
  __builtin_amdgcn_wmma_f32_16x16x32_f16(false, (A), false, (B), (short)0, (C), false, false)

// ---- async global->LDS copy (CDNA5 path, tracked by ASYNCcnt) -------------
__device__ __forceinline__ void async_b128(unsigned ldsAddr,
                                           unsigned long long gbase,
                                           unsigned byteOff) {
  asm volatile("global_load_async_to_lds_b128 %0, %1, %2"
               :: "v"(ldsAddr), "v"(byteOff), "s"(gbase)
               : "memory");
}
__device__ __forceinline__ void wait_async8() {
  asm volatile("s_wait_asynccnt 0x8" ::: "memory");
}
__device__ __forceinline__ void wait_async0() {
  asm volatile("s_wait_asynccnt 0x0" ::: "memory");
}

// issue one 128x64-half tile pair (A and B), 4 b128 per matrix per thread
__device__ __forceinline__ void issue_tile(unsigned ldsA, unsigned ldsB,
                                           unsigned long long Ag,
                                           unsigned long long Bg,
                                           unsigned aOff, unsigned bOff) {
  async_b128(ldsA,      Ag, aOff);
  async_b128(ldsA + 16, Ag, aOff + 16);
  async_b128(ldsA + 32, Ag, aOff + 32);
  async_b128(ldsA + 48, Ag, aOff + 48);
  async_b128(ldsB,      Bg, bOff);
  async_b128(ldsB + 16, Bg, bOff + 16);
  async_b128(ldsB + 32, Bg, bOff + 32);
  async_b128(ldsB + 48, Bg, bOff + 48);
}

// A-fragment (16x32 f16, M x K) at k-subchunk ks (0/32): lane m = lane&15.
__device__ __forceinline__ v16h a_frag(const _Float16* tile, int row0, int ks,
                                       int lane) {
  const int m = row0 + (lane & 15);
  const int h = (lane >> 4) * 8;
  const _Float16* p = tile + m * LDSROW + ks;
  v16h a;
  ((v8h*)&a)[0] = *(const v8h*)(p + h);
  ((v8h*)&a)[1] = *(const v8h*)(p + 16 + h);
  return a;
}

// B-fragment (32x16 f16, K x N) from row-major B^T rows, k-subchunk ks.
__device__ __forceinline__ v16h b_frag(const _Float16* tile, int row0, int ks,
                                       int lane) {
  const int n = row0 + (lane & 15);
  const int h = (lane >> 4) * 8;
  const _Float16* p = tile + n * LDSROW + ks;
  v16h b;
  ((v8h*)&b)[0] = *(const v8h*)(p + h);
  ((v8h*)&b)[1] = *(const v8h*)(p + 16 + h);
  return b;
}

__device__ __forceinline__ void store_tile_f16(_Float16* __restrict__ outp, int ldc,
                                               int r0, int c0, int lane,
                                               const v8f& acc, float scale) {
  const int col = c0 + (lane & 15);
  const int row = r0 + 8 * (lane >> 4);
#pragma unroll
  for (int r = 0; r < 8; ++r)
    outp[(size_t)(row + r) * ldc + col] = (_Float16)(acc[r] * scale);
}

// ---------------------------------------------------------------------------
// K1: column softmax of wv (f32 -> f16): wvS[j,k] = exp(wv[j,k]-mx_k)/sum_k
// ---------------------------------------------------------------------------
__global__ __launch_bounds__(256) void col_softmax(const float* __restrict__ wv,
                                                   _Float16* __restrict__ wvS) {
  const int col = blockIdx.x * 256 + threadIdx.x;
  float mx = -1e30f;
  for (int j = 0; j < 1024; ++j) mx = fmaxf(mx, wv[(size_t)j * 1024 + col]);
  float sum = 0.f;
  for (int j = 0; j < 1024; ++j)
    sum += exp2f((wv[(size_t)j * 1024 + col] - mx) * L2E);
  const float inv = 1.0f / sum;
  for (int j = 0; j < 1024; ++j)
    wvS[(size_t)j * 1024 + col] =
        (_Float16)(exp2f((wv[(size_t)j * 1024 + col] - mx) * L2E) * inv);
}

// ---------------------------------------------------------------------------
// K2: 1024x1024 transpose + f32->f16 (wq, wk)
// ---------------------------------------------------------------------------
__global__ __launch_bounds__(256) void transpose_w(const float* __restrict__ in,
                                                   _Float16* __restrict__ outp) {
  __shared__ float tile[32][33];
  const int tx = threadIdx.x, ty = threadIdx.y;
  int x = blockIdx.x * 32 + tx;
  int y = blockIdx.y * 32 + ty;
#pragma unroll
  for (int i = 0; i < 32; i += 8)
    tile[ty + i][tx] = in[(size_t)(y + i) * 1024 + x];
  __syncthreads();
  x = blockIdx.y * 32 + tx;
  y = blockIdx.x * 32 + ty;
#pragma unroll
  for (int i = 0; i < 32; i += 8)
    outp[(size_t)(y + i) * 1024 + x] = (_Float16)tile[tx][ty + i];
}

// ---------------------------------------------------------------------------
// K3: bulk f32 -> f16 convert (x -> xh), 8 elems/thread
// ---------------------------------------------------------------------------
__global__ __launch_bounds__(256) void cvt_f32_f16(const float* __restrict__ in,
                                                   _Float16* __restrict__ outp) {
  const size_t i = ((size_t)blockIdx.x * 256 + threadIdx.x) * 8;
  const v4f a = ((const v4f*)(in + i))[0];
  const v4f b = ((const v4f*)(in + i))[1];
  v8h h;
#pragma unroll
  for (int k = 0; k < 4; ++k) { h[k] = (_Float16)a[k]; h[4 + k] = (_Float16)b[k]; }
  *(v8h*)(outp + i) = h;
}

// ---------------------------------------------------------------------------
// K4: generic f16 GEMM  C = scale * A @ B^T  (A, B row-major f16)
// block tile 128x128, 8 waves (wave tile 32x64), k-step 64 (2 WMMA subchunks),
// double-buffered LDS fed by global_load_async_to_lds_b128; last tile peeled
// so the steady-state loop has constant ASYNCcnt waits.
// ---------------------------------------------------------------------------
__global__ __launch_bounds__(256) void gemm_ff(
    const _Float16* __restrict__ A, const _Float16* __restrict__ B,
    _Float16* __restrict__ C, int K, int lda, int ldb, int ldc,
    unsigned long long sA, unsigned long long sB, unsigned long long sC,
    float scale) {
  __shared__ alignas(16) _Float16 As[2][128 * LDSROW];
  __shared__ alignas(16) _Float16 Bs[2][128 * LDSROW];
  const int t = threadIdx.x, lane = t & 31, wave = t >> 5;
  const int wm = (wave & 3) * 32, wn = (wave >> 2) * 64;
  const int mBase = blockIdx.x * 128, nBase = blockIdx.y * 128;
  const int lr = t >> 1, lc = (t & 1) * 32;

  const unsigned long long Ag = (unsigned long long)(A + (size_t)blockIdx.z * sA);
  const unsigned long long Bg = (unsigned long long)(B + (size_t)blockIdx.z * sB);
  _Float16* Cb = C + (size_t)blockIdx.z * sC;

  const unsigned ldsA = (unsigned)(unsigned long long)&As[0][lr * LDSROW + lc];
  const unsigned ldsB = (unsigned)(unsigned long long)&Bs[0][lr * LDSROW + lc];
  const unsigned aOff = (unsigned)(((mBase + lr) * lda + lc) * 2);
  const unsigned bOff = (unsigned)(((nBase + lr) * ldb + lc) * 2);

  v8f acc[2][4] = {};

  auto compute = [&](const _Float16* at, const _Float16* bt) {
#pragma unroll
    for (int ks = 0; ks < 64; ks += 32) {
      v16h bf[4];
#pragma unroll
      for (int bn = 0; bn < 4; ++bn) bf[bn] = b_frag(bt, wn + bn * 16, ks, lane);
      const v16h a0 = a_frag(at, wm, ks, lane);
      const v16h a1 = a_frag(at, wm + 16, ks, lane);
#pragma unroll
      for (int bn = 0; bn < 4; ++bn) {
        acc[0][bn] = WMMA_F16(a0, bf[bn], acc[0][bn]);
        acc[1][bn] = WMMA_F16(a1, bf[bn], acc[1][bn]);
      }
    }
  };

  // prologue: tile 0 -> buffer 0
  issue_tile(ldsA, ldsB, Ag, Bg, aOff, bOff);

  int buf = 0;
  for (int kc = 0; kc < K - 64; kc += 64) {
    __syncthreads();                               // buf^1 free for overwrite
    const unsigned nb = (unsigned)(buf ^ 1) * (unsigned)TILEB;
    const unsigned ko = (unsigned)(kc + 64) * 2u;
    issue_tile(ldsA + nb, ldsB + nb, Ag, Bg, aOff + ko, bOff + ko);
    wait_async8();                                 // own tile-kc copies done
    __syncthreads();                               // tile kc visible to all
    compute(As[buf], Bs[buf]);
    buf ^= 1;
  }
  // peeled last tile
  wait_async0();
  __syncthreads();
  compute(As[buf], Bs[buf]);

#pragma unroll
  for (int am = 0; am < 2; ++am)
#pragma unroll
    for (int bn = 0; bn < 4; ++bn)
      store_tile_f16(Cb, ldc, mBase + wm + am * 16, nBase + wn + bn * 16,
                     lane, acc[am][bn], scale);
}

// ---------------------------------------------------------------------------
// K5: column softmax stats of S: c2[b,m] = 10 - log2(sum_n exp(s-mx)) - mx*L2E
// ---------------------------------------------------------------------------
__global__ __launch_bounds__(256) void col_stats(const _Float16* __restrict__ S,
                                                 float* __restrict__ c2) {
  const int b = blockIdx.y;
  const _Float16* Sb = S + (size_t)b * 2048 * 2048;
  const int c = threadIdx.x & 63;
  const int col = blockIdx.x * 64 + c;
  const int rg = threadIdx.x >> 6;
  __shared__ float red[4][64];

  float mx = -1e30f;
  for (int n = rg; n < 2048; n += 4)
    mx = fmaxf(mx, (float)Sb[(size_t)n * 2048 + col]);
  red[rg][c] = mx;
  __syncthreads();
  if (rg == 0)
    red[0][c] = fmaxf(fmaxf(red[0][c], red[1][c]), fmaxf(red[2][c], red[3][c]));
  __syncthreads();
  mx = red[0][c];
  __syncthreads();

  float sum = 0.f;
  for (int n = rg; n < 2048; n += 4)
    sum += exp2f(((float)Sb[(size_t)n * 2048 + col] - mx) * L2E);
  red[rg][c] = sum;
  __syncthreads();
  if (rg == 0) {
    const float s = red[0][c] + red[1][c] + red[2][c] + red[3][c];
    c2[(size_t)b * 2048 + col] = 10.0f - log2f(s) - mx * L2E;
  }
}

// ---------------------------------------------------------------------------
// K6: batched f16 transpose: Vu[b,2048,1024] -> VT[b,1024,2048]
// ---------------------------------------------------------------------------
__global__ __launch_bounds__(256) void transpose_v(const _Float16* __restrict__ in,
                                                   _Float16* __restrict__ outp) {
  const int b = blockIdx.z;
  const _Float16* ib = in + (size_t)b * 2048 * 1024;
  _Float16* ob = outp + (size_t)b * 2048 * 1024;
  __shared__ _Float16 tile[32][33];
  const int tx = threadIdx.x, ty = threadIdx.y;
  int x = blockIdx.x * 32 + tx;
  int y = blockIdx.y * 32 + ty;
#pragma unroll
  for (int i = 0; i < 32; i += 8)
    tile[ty + i][tx] = ib[(size_t)(y + i) * 1024 + x];
  __syncthreads();
  x = blockIdx.y * 32 + tx;
  y = blockIdx.x * 32 + ty;
#pragma unroll
  for (int i = 0; i < 32; i += 8)
    ob[(size_t)(y + i) * 2048 + x] = tile[tx][ty + i];
}

// ---------------------------------------------------------------------------
// K7: Out[b,n,j] = ( (1/1024)*sum_m exp2(S*L2E + c2[m]) * Vu[m,j] - Vu[n,j] )/1024
//     P built on the fly; same async double-buffered, last-tile-peeled skeleton.
// ---------------------------------------------------------------------------
__device__ __forceinline__ v16h exp_frag(v16h a, const float* cv) {
  v16h r;
#pragma unroll
  for (int j = 0; j < 16; ++j) {
    const float s = (float)a[j];
    r[j] = (_Float16)exp2f(fmaf(s, L2E, cv[j]));
  }
  return r;
}

__global__ __launch_bounds__(256) void gemm_pv(const _Float16* __restrict__ S,
                                               const float* __restrict__ c2,
                                               const _Float16* __restrict__ VT,
                                               const _Float16* __restrict__ Vu,
                                               float* __restrict__ Out) {
  __shared__ alignas(16) _Float16 As[2][128 * LDSROW];
  __shared__ alignas(16) _Float16 Bs[2][128 * LDSROW];
  const int b = blockIdx.z;
  const float* c2b = c2 + (size_t)b * 2048;
  const _Float16* Vub = Vu + (size_t)b * 2048 * 1024;
  float* Outb = Out + (size_t)b * 2048 * 1024;

  const int t = threadIdx.x, lane = t & 31, wave = t >> 5;
  const int wm = (wave & 3) * 32, wn = (wave >> 2) * 64;
  const int mBase = blockIdx.x * 128, nBase = blockIdx.y * 128;
  const int lr = t >> 1, lc = (t & 1) * 32;

  const unsigned long long Sg =
      (unsigned long long)(S + (size_t)b * 2048 * 2048);
  const unsigned long long Vg =
      (unsigned long long)(VT + (size_t)b * 2048 * 1024);

  const unsigned ldsA = (unsigned)(unsigned long long)&As[0][lr * LDSROW + lc];
  const unsigned ldsB = (unsigned)(unsigned long long)&Bs[0][lr * LDSROW + lc];
  const unsigned aOff = (unsigned)(((mBase + lr) * 2048 + lc) * 2);
  const unsigned bOff = (unsigned)(((nBase + lr) * 2048 + lc) * 2);

  v8f acc[2][4] = {};

  auto compute = [&](const _Float16* at, const _Float16* bt, int kcAbs) {
#pragma unroll
    for (int ks = 0; ks < 64; ks += 32) {
      // per-lane softmax constants matching the A-frag k pattern
      const int h8 = (lane >> 4) * 8;
      const float* cp = c2b + kcAbs + ks + h8;
      const v4f c0 = ((const v4f*)cp)[0], c1 = ((const v4f*)cp)[1];
      const float* cp2 = c2b + kcAbs + ks + 16 + h8;
      const v4f c2a = ((const v4f*)cp2)[0], c2bv = ((const v4f*)cp2)[1];
      float cv[16];
#pragma unroll
      for (int i = 0; i < 4; ++i) {
        cv[i] = c0[i]; cv[4 + i] = c1[i]; cv[8 + i] = c2a[i]; cv[12 + i] = c2bv[i];
      }
      v16h bf[4];
#pragma unroll
      for (int bn = 0; bn < 4; ++bn) bf[bn] = b_frag(bt, wn + bn * 16, ks, lane);
      const v16h a0 = exp_frag(a_frag(at, wm, ks, lane), cv);
      const v16h a1 = exp_frag(a_frag(at, wm + 16, ks, lane), cv);
#pragma unroll
      for (int bn = 0; bn < 4; ++bn) {
        acc[0][bn] = WMMA_F16(a0, bf[bn], acc[0][bn]);
        acc[1][bn] = WMMA_F16(a1, bf[bn], acc[1][bn]);
      }
    }
  };

  issue_tile(ldsA, ldsB, Sg, Vg, aOff, bOff);

  int buf = 0;
  for (int kc = 0; kc < 2048 - 64; kc += 64) {
    __syncthreads();
    const unsigned nb = (unsigned)(buf ^ 1) * (unsigned)TILEB;
    const unsigned ko = (unsigned)(kc + 64) * 2u;
    issue_tile(ldsA + nb, ldsB + nb, Sg, Vg, aOff + ko, bOff + ko);
    wait_async8();
    __syncthreads();
    compute(As[buf], Bs[buf], kc);
    buf ^= 1;
  }
  wait_async0();
  __syncthreads();
  compute(As[buf], Bs[buf], 2048 - 64);

#pragma unroll
  for (int am = 0; am < 2; ++am)
#pragma unroll
    for (int bn = 0; bn < 4; ++bn) {
      const int col = nBase + wn + bn * 16 + (lane & 15);
      const int row = mBase + wm + am * 16 + 8 * (lane >> 4);
#pragma unroll
      for (int r = 0; r < 8; ++r) {
        const size_t idx = (size_t)(row + r) * 1024 + col;
        Outb[idx] = (acc[am][bn][r] * (1.0f / 1024.0f) - (float)Vub[idx]) *
                    (1.0f / 1024.0f);
      }
    }
}

// ---------------------------------------------------------------------------
// Host launcher
// ---------------------------------------------------------------------------
extern "C" void kernel_launch(void* const* d_in, const int* in_sizes, int n_in,
                              void* d_out, int out_size, void* d_ws, size_t ws_size,
                              hipStream_t stream) {
  (void)in_sizes; (void)n_in; (void)out_size; (void)ws_size;
  const float* x  = (const float*)d_in[0];   // [4,2048,1024]
  const float* wk = (const float*)d_in[1];   // [1024,1024]
  const float* wq = (const float*)d_in[2];
  const float* wv = (const float*)d_in[3];
  float* out = (float*)d_out;                // [4,2048,1024] f32

  char* ws = (char*)d_ws;
  _Float16* wvS  = (_Float16*)(ws);                          //  2 MB
  _Float16* wqT  = (_Float16*)(ws + (size_t)(2   << 20));    //  2 MB
  _Float16* wkT  = (_Float16*)(ws + (size_t)(4   << 20));    //  2 MB
  _Float16* xh   = (_Float16*)(ws + (size_t)(6   << 20));    // 16 MB
  _Float16* Qm   = (_Float16*)(ws + (size_t)(22  << 20));    // 16 MB
  _Float16* Km   = (_Float16*)(ws + (size_t)(38  << 20));    // 16 MB
  _Float16* Vu   = (_Float16*)(ws + (size_t)(54  << 20));    // 16 MB
  _Float16* VT   = (_Float16*)(ws + (size_t)(70  << 20));    // 16 MB
  _Float16* Smat = (_Float16*)(ws + (size_t)(86  << 20));    // 32 MB
  float*    c2   = (float*)   (ws + (size_t)(118 << 20));    // 32 KB

  col_softmax<<<4, 256, 0, stream>>>(wv, wvS);
  transpose_w<<<dim3(32, 32), dim3(32, 8), 0, stream>>>(wq, wqT);
  transpose_w<<<dim3(32, 32), dim3(32, 8), 0, stream>>>(wk, wkT);
  cvt_f32_f16<<<4096, 256, 0, stream>>>(x, xh);

  // Q = x@wq, K = x@wk, V(unscaled) = x@softmax_col(wv)
  gemm_ff<<<dim3(64, 8, 1), 256, 0, stream>>>(xh, wqT, Qm, 1024, 1024, 1024, 1024,
                                              0ull, 0ull, 0ull, 1.0f);
  gemm_ff<<<dim3(64, 8, 1), 256, 0, stream>>>(xh, wkT, Km, 1024, 1024, 1024, 1024,
                                              0ull, 0ull, 0ull, 1.0f);
  gemm_ff<<<dim3(64, 8, 1), 256, 0, stream>>>(xh, wvS, Vu, 1024, 1024, 1024, 1024,
                                              0ull, 0ull, 0ull, 1.0f);

  // S[b,n,m] = (2048^-0.5) * Q . K
  gemm_ff<<<dim3(16, 16, 4), 256, 0, stream>>>(
      Qm, Km, Smat, 1024, 1024, 1024, 2048,
      (unsigned long long)2048 * 1024, (unsigned long long)2048 * 1024,
      (unsigned long long)2048 * 2048, 0.02209708691207961f);

  col_stats<<<dim3(32, 4), 256, 0, stream>>>(Smat, c2);
  transpose_v<<<dim3(32, 64, 4), dim3(32, 8), 0, stream>>>(Vu, VT);

  gemm_pv<<<dim3(16, 8, 4), 256, 0, stream>>>(Smat, c2, VT, Vu, out);
}
